// NequIP_2113123909659
// MI455X (gfx1250) — matrix operand order repaired and checked by reading, not compiled, not run on hardware
//
#include <hip/hip_runtime.h>

#define NN 1024
#define NE 8192
#define NG 8
#define HD 32

typedef __attribute__((ext_vector_type(16))) __bf16 v16bf;
typedef __attribute__((ext_vector_type(8)))  float  v8f;

// ---------------- constants (real spherical harmonics / CG couplings) -------
#define CY1  0.4886025119f
#define KB00 0.28209479f
#define C10  0.57735027f
#define E111 0.70710678f
#define KSA  0.6324555f
#define KSB  0.1825742f
#define KSC  0.3162278f
#define C2A  1.0925484f
#define C2B  0.31539157f
#define C2C  0.54627422f

__host__ __device__ constexpr int epsv(int a, int b, int c) {
  return (a == b || b == c || a == c) ? 0 : (((b - a + 3) % 3 == 1) ? 1 : -1);
}

// pair types: 0=(di0,do0) 1=(di0,do1) 2=(di1,do0) 3=(di1,do1)
template<int PT>
__device__ inline void computeBasis(const float* d, float* bas) {
  const float x = d[0], y = d[1], z = d[2];
  const float Y1[3] = {CY1 * y, CY1 * z, CY1 * x};
  if (PT == 1) {
    bas[0] = Y1[0]; bas[1] = Y1[1]; bas[2] = Y1[2];
  } else if (PT == 2) {
    bas[0] = C10 * Y1[0]; bas[1] = C10 * Y1[1]; bas[2] = C10 * Y1[2];
  } else if (PT == 3) {
    const int AX[3] = {1, 2, 0};
    float Y2[5] = {C2A * x * y, C2A * y * z, C2B * (3.f * z * z - 1.f),
                   C2A * x * z, C2C * (x * x - y * y)};
    float S[3][3];
    S[0][0] =  0.5f * KSA * Y2[4] - KSB * Y2[2];
    S[1][1] = -0.5f * KSA * Y2[4] - KSB * Y2[2];
    S[2][2] =  2.f * KSB * Y2[2];
    S[0][1] = S[1][0] = KSC * Y2[0];
    S[1][2] = S[2][1] = KSC * Y2[1];
    S[0][2] = S[2][0] = KSC * Y2[3];
    for (int u = 0; u < 3; ++u)
      for (int v = 0; v < 3; ++v) {
        bas[(u * 3 + v) * 3 + 0] = (u == v) ? KB00 : 0.f;
        float e1 = 0.f;
        for (int j = 0; j < 3; ++j) e1 += (float)epsv(AX[v], AX[j], AX[u]) * Y1[j];
        bas[(u * 3 + v) * 3 + 1] = E111 * e1;
        bas[(u * 3 + v) * 3 + 2] = S[AX[u]][AX[v]];
      }
  }
}

// A-fragment (16x32 bf16, M = edge row, K = hidden) per ISA VGPR layout
__device__ inline v16bf load_h2_frag(const __bf16* h2bf, int e0, int lane) {
  v16bf a;
  const int e = e0 + (lane & 15);
#pragma unroll
  for (int i = 0; i < 16; ++i) {
    int vg = i >> 1, hf = i & 1;
    int K = ((vg & 3) * 2 + hf) + ((lane >> 4) ? 8 : 0) + ((vg >> 2) ? 16 : 0);
    a[i] = h2bf[(size_t)e * HD + K];
  }
  return a;
}

// ------------------------------ simple kernels ------------------------------
__global__ void k_geom(const float* pos, const int* src, const int* dst,
                       float* r, float* dirs, float* cnt) {
  int e = blockIdx.x * blockDim.x + threadIdx.x;
  if (e >= NE) return;
  int s = src[e], d = dst[e];
  float vx = pos[d * 3 + 0] - pos[s * 3 + 0];
  float vy = pos[d * 3 + 1] - pos[s * 3 + 1];
  float vz = pos[d * 3 + 2] - pos[s * 3 + 2];
  float rr = sqrtf(vx * vx + vy * vy + vz * vz + 1e-12f);
  r[e] = rr;
  dirs[e * 3 + 0] = vx / rr; dirs[e * 3 + 1] = vy / rr; dirs[e * 3 + 2] = vz / rr;
  atomicAdd(&cnt[d], 1.f);
}

__global__ void k_invdeg(const float* cnt, float* invd) {
  int n = blockIdx.x * blockDim.x + threadIdx.x;
  if (n < NN) invd[n] = 1.f / fmaxf(cnt[n], 1.f);
}

__global__ void k_embed(const int* feats, const float* emb, float* h0) {
  int i = blockIdx.x * blockDim.x + threadIdx.x;
  if (i < NN * HD) h0[i] = emb[feats[i >> 5] * HD + (i & 31)];
}

__global__ void k_w3prep(const float* W3, __bf16* w3bf, __bf16* w3t, int cols) {
  int i = blockIdx.x * blockDim.x + threadIdx.x;
  if (i >= 32 * cols) return;
  int m = i / cols, c = i - m * cols;
  float v = W3[i];
  w3bf[i] = (__bf16)v;
  w3t[(size_t)c * 32 + m] = (__bf16)v;
}

__global__ void k_radial_fwd(const float* r, const float* W1, const float* b1,
                             const float* W2, const float* b2,
                             float* h1o, float* h2o, __bf16* h2bf) {
  int e = blockIdx.x * 8 + (threadIdx.x >> 5);
  int m = threadIdx.x & 31;
  float rr = r[e];
  float a1 = fmaxf(rr * W1[m] + b1[m], 0.f);
  float acc = b2[m];
  for (int k = 0; k < 32; ++k) acc += __shfl(a1, k) * W2[k * 32 + m];
  float h2 = fmaxf(acc, 0.f);
  h1o[(size_t)e * HD + m] = a1;
  h2o[(size_t)e * HD + m] = h2;
  h2bf[(size_t)e * HD + m] = (__bf16)h2;
}

__global__ void k_radial_bwd(const float* gh2in, const float* h1, const float* h2,
                             const float* W1, const float* W2, float* g_r) {
  int e = blockIdx.x * 8 + (threadIdx.x >> 5);
  int m = threadIdx.x & 31;
  float gh2 = gh2in[(size_t)e * HD + m] * (h2[(size_t)e * HD + m] > 0.f ? 1.f : 0.f);
  float acc = 0.f;  // lane acts as k
  for (int mm = 0; mm < 32; ++mm) acc += __shfl(gh2, mm) * W2[m * 32 + mm];
  float gh1 = acc * (h1[(size_t)e * HD + m] > 0.f ? 1.f : 0.f);
  float part = gh1 * W1[m];
  for (int off = 16; off; off >>= 1) part += __shfl_xor(part, off);
  if (m == 0) g_r[e] += part;
}

__global__ void k_node_upd(const float* acc, const float* invd, const float* s,
                           const float* hin, float* hout, int U) {
  int i = blockIdx.x * blockDim.x + threadIdx.x;
  if (i >= NN * HD * U) return;
  int n = i / (HD * U), rem = i - n * HD * U;
  int o = rem / U, u = rem - o * U;
  float v = acc[i] * invd[n];
  if (s) {
    const float* hn = hin + (size_t)n * HD * U;
    float t = 0.f;
    for (int c = 0; c < 32; ++c) t += s[o * 32 + c] * hn[c * U + u];
    v += t;
  }
  hout[i] = v;
}

__global__ void k_node_upd_bwd(const float* gout, const float* invd, const float* s,
                               float* gacc, float* ghin, int U) {
  int i = blockIdx.x * blockDim.x + threadIdx.x;
  if (i >= NN * HD * U) return;
  int n = i / (HD * U), rem = i - n * HD * U;
  int a = rem / U, u = rem - a * U;   // a: o for gacc, c for ghin
  gacc[i] = gout[i] * invd[n];
  if (s && ghin) {
    const float* gn = gout + (size_t)n * HD * U;
    float t = 0.f;
    for (int o2 = 0; o2 < 32; ++o2) t += s[o2 * 32 + a] * gn[o2 * U + u];
    ghin[i] += t;
  }
}

__global__ __launch_bounds__(32)
void k_readout_fwd(const float* hD, const float* W1, const float* b1,
                   const float* W2, const float* b2, const int* gid, float* total) {
  int n = blockIdx.x, m = threadIdx.x;
  float sv = hD[n * HD + m];
  float pre = b1[m];
  for (int c = 0; c < 32; ++c) pre += __shfl(sv, c) * W1[c * 32 + m];
  float sig = 1.f / (1.f + __expf(-pre));
  float part = pre * sig * W2[m];
  for (int off = 16; off; off >>= 1) part += __shfl_xor(part, off);
  if (m == 0) atomicAdd(&total[gid[n]], part + b2[0]);
}

__global__ __launch_bounds__(32)
void k_readout_bwd(const float* hD, const float* W1, const float* b1,
                   const float* W2, float* ghD) {
  int n = blockIdx.x, m = threadIdx.x;
  float sv = hD[n * HD + m];
  float pre = b1[m];
  for (int c = 0; c < 32; ++c) pre += __shfl(sv, c) * W1[c * 32 + m];
  float sig = 1.f / (1.f + __expf(-pre));
  float dsl = sig * (1.f + pre * (1.f - sig));
  float gpre = W2[m] * dsl;
  float gs = 0.f;  // lane acts as c
  for (int mm = 0; mm < 32; ++mm) gs += __shfl(gpre, mm) * W1[m * 32 + mm];
  ghD[n * HD + m] = gs;
}

// ------------------- fused WMMA message kernels (fwd) -----------------------
template<int NF, int VDI, int UDO, int PT>
__global__ __launch_bounds__(32)
void k_msg_fwd(const int* __restrict__ src, const int* __restrict__ dst,
               const float* __restrict__ dirs, const float* __restrict__ hin,
               const __bf16* __restrict__ h2bf, const __bf16* __restrict__ w3bf,
               const float* __restrict__ b3, float* __restrict__ tmpg,
               float* __restrict__ acc) {
  __shared__ float Rb[16][8][32];
  __shared__ float featL[16][32][VDI];
  __shared__ float msgL[16][32][UDO];
  __shared__ float basL[16][UDO * VDI * NF];
  const int lane = threadIdx.x;
  const int e0 = blockIdx.x << 4;
  for (int i = lane; i < 16 * 32 * VDI; i += 32) {
    int e = i / (32 * VDI), rm = i - e * 32 * VDI;
    featL[e][rm / VDI][rm % VDI] = hin[(size_t)src[e0 + e] * (32 * VDI) + rm];
  }
  for (int i = lane; i < 16 * 32 * UDO; i += 32) {
    int e = i / (32 * UDO), rm = i - e * 32 * UDO;
    msgL[e][rm / UDO][rm % UDO] = 0.f;
  }
  if (PT != 0 && lane < 16) computeBasis<PT>(dirs + (size_t)(e0 + lane) * 3, basL[lane]);
  __syncthreads();
  v16bf a = load_h2_frag(h2bf, e0, lane);
  const int COLS = NF * 1024;
  for (int f = 0; f < NF; ++f) {
    for (int ob = 0; ob < 4; ++ob) {
      // prefetch next weight block (global_prefetch_b8) while this one computes
      {
        int nf2 = f, nob = ob + 1;
        if (nob == 4) { nob = 0; nf2 = f + 1; }
        if (nf2 < NF) {
          const __bf16* nxt = w3bf + (size_t)lane * COLS + nf2 * 1024 + nob * 256;
          __builtin_prefetch(nxt, 0, 3);
          __builtin_prefetch(nxt + 128, 0, 3);
        }
      }
      for (int tt = 0; tt < 16; ++tt) {
        const int t = ob * 16 + tt;
        v16bf b;
        const __bf16* wr = w3bf + f * 1024 + t * 16 + (lane & 15);
#pragma unroll
        for (int i = 0; i < 16; ++i) b[i] = wr[(size_t)((i & 7) + ((lane >> 4) << 3) + ((i >> 3) << 4)) * COLS];
        v8f c = {0.f, 0.f, 0.f, 0.f, 0.f, 0.f, 0.f, 0.f};
        c = __builtin_amdgcn_wmma_f32_16x16x32_bf16(false, a, false, b, (short)0, c, false, false);
        const int coli = t * 16 + (lane & 15);
        const int o = coli >> 5, cc = coli & 31;
        const float bias = b3[f * 1024 + coli];
#pragma unroll
        for (int i = 0; i < 8; ++i) {
          int er = i + ((lane >> 4) << 3);
          Rb[er][o - ob * 8][cc] = c[i] + bias;
        }
      }
      __syncthreads();
      {
        const int e = lane & 15, hf = lane >> 4;
        for (int oo = 0; oo < 4; ++oo) {
          const int o = ob * 8 + hf * 4 + oo;
          float tv[VDI];
#pragma unroll
          for (int v = 0; v < VDI; ++v) tv[v] = 0.f;
          for (int cc = 0; cc < 32; ++cc) {
            float rv = Rb[e][o - ob * 8][cc];
#pragma unroll
            for (int v = 0; v < VDI; ++v) tv[v] += rv * featL[e][cc][v];
          }
#pragma unroll
          for (int v = 0; v < VDI; ++v)
            tmpg[(((size_t)(e0 + e) * NF + f) * 32 + o) * VDI + v] = tv[v];
          if (PT == 0) {
            msgL[e][o][0] += tv[0] * KB00;
          } else {
#pragma unroll
            for (int u = 0; u < UDO; ++u) {
              float m = 0.f;
#pragma unroll
              for (int v = 0; v < VDI; ++v) m += tv[v] * basL[e][(u * VDI + v) * NF + f];
              msgL[e][o][u] += m;
            }
          }
        }
      }
      __syncthreads();
    }
  }
  for (int i = lane; i < 16 * 32 * UDO; i += 32) {
    int e = i / (32 * UDO), rm = i - e * 32 * UDO;
    atomicAdd(&acc[(size_t)dst[e0 + e] * (32 * UDO) + rm], msgL[e][rm / UDO][rm % UDO]);
  }
}

// ------------------- fused WMMA message kernels (bwd) -----------------------
template<int NF, int VDI, int UDO, int PT>
__global__ __launch_bounds__(32)
void k_msg_bwd(const int* __restrict__ src, const int* __restrict__ dst,
               const float* __restrict__ dirs, const float* __restrict__ hin,
               const __bf16* __restrict__ h2bf, const __bf16* __restrict__ w3bf,
               const __bf16* __restrict__ w3t, const float* __restrict__ b3,
               const float* __restrict__ tmpg, const float* __restrict__ gacc,
               float* __restrict__ ghin, float* __restrict__ gh2,
               float* __restrict__ gbas) {
  __shared__ float Rb[16][8][32];
  __shared__ float gRb[16][8][32];
  __shared__ float featL[16][32][VDI];
  __shared__ float gfeatL[16][32][VDI];
  __shared__ float gmsgL[16][32][UDO];
  __shared__ float gtmpL[16][8][VDI];
  __shared__ float basL[16][UDO * VDI * NF];
  __shared__ float gbasL[16][UDO * VDI * NF];
  const int lane = threadIdx.x;
  const int e0 = blockIdx.x << 4;
  for (int i = lane; i < 16 * 32 * VDI; i += 32) {
    int e = i / (32 * VDI), rm = i - e * 32 * VDI;
    featL[e][rm / VDI][rm % VDI] = hin[(size_t)src[e0 + e] * (32 * VDI) + rm];
    gfeatL[e][rm / VDI][rm % VDI] = 0.f;
  }
  for (int i = lane; i < 16 * 32 * UDO; i += 32) {
    int e = i / (32 * UDO), rm = i - e * 32 * UDO;
    gmsgL[e][rm / UDO][rm % UDO] = gacc[(size_t)dst[e0 + e] * (32 * UDO) + rm];
  }
  for (int i = lane; i < 16 * (UDO * VDI * NF); i += 32)
    gbasL[i / (UDO * VDI * NF)][i % (UDO * VDI * NF)] = 0.f;
  if (PT != 0 && lane < 16) computeBasis<PT>(dirs + (size_t)(e0 + lane) * 3, basL[lane]);
  __syncthreads();
  v16bf a = load_h2_frag(h2bf, e0, lane);
  v8f cg0 = {0.f, 0.f, 0.f, 0.f, 0.f, 0.f, 0.f, 0.f};
  v8f cg1 = {0.f, 0.f, 0.f, 0.f, 0.f, 0.f, 0.f, 0.f};
  const int COLS = NF * 1024;
  for (int f = 0; f < NF; ++f) {
    float pb[UDO * VDI];
#pragma unroll
    for (int k = 0; k < UDO * VDI; ++k) pb[k] = 0.f;
    for (int ob = 0; ob < 4; ++ob) {
      // prefetch next W3 / W3^T blocks for this wave's lanes
      {
        int nf2 = f, nob = ob + 1;
        if (nob == 4) { nob = 0; nf2 = f + 1; }
        if (nf2 < NF) {
          const __bf16* nxt = w3bf + (size_t)lane * COLS + nf2 * 1024 + nob * 256;
          __builtin_prefetch(nxt, 0, 3);
          __builtin_prefetch(nxt + 128, 0, 3);
          const __bf16* nxtT = w3t + (size_t)(nf2 * 1024 + nob * 256 + lane) * HD;
          __builtin_prefetch(nxtT, 0, 3);
        }
      }
      // recompute R tile (with bias) via WMMA
      for (int tt = 0; tt < 16; ++tt) {
        const int t = ob * 16 + tt;
        v16bf b;
        const __bf16* wr = w3bf + f * 1024 + t * 16 + (lane & 15);
#pragma unroll
        for (int i = 0; i < 16; ++i) b[i] = wr[(size_t)((i & 7) + ((lane >> 4) << 3) + ((i >> 3) << 4)) * COLS];
        v8f c = {0.f, 0.f, 0.f, 0.f, 0.f, 0.f, 0.f, 0.f};
        c = __builtin_amdgcn_wmma_f32_16x16x32_bf16(false, a, false, b, (short)0, c, false, false);
        const int coli = t * 16 + (lane & 15);
        const int o = coli >> 5, cc = coli & 31;
        const float bias = b3[f * 1024 + coli];
#pragma unroll
        for (int i = 0; i < 8; ++i) {
          int er = i + ((lane >> 4) << 3);
          Rb[er][o - ob * 8][cc] = c[i] + bias;
        }
      }
      __syncthreads();
      {
        const int e = lane & 15, hf = lane >> 4;
        for (int oo = 0; oo < 4; ++oo) {
          const int o = ob * 8 + hf * 4 + oo;
          float gm[UDO];
#pragma unroll
          for (int u = 0; u < UDO; ++u) gm[u] = gmsgL[e][o][u];
#pragma unroll
          for (int v = 0; v < VDI; ++v) {
            float g = 0.f;
            if (PT == 0) {
              g = gm[0] * KB00;
            } else {
              float tgv = tmpg[(((size_t)(e0 + e) * NF + f) * 32 + o) * VDI + v];
#pragma unroll
              for (int u = 0; u < UDO; ++u) {
                g += gm[u] * basL[e][(u * VDI + v) * NF + f];
                pb[u * VDI + v] += gm[u] * tgv;
              }
            }
            gtmpL[e][o - ob * 8][v] = g;
          }
        }
      }
      __syncthreads();
      {
        const int e = lane & 15, hf = lane >> 4;
        for (int oo = 0; oo < 4; ++oo) {
          const int ol = hf * 4 + oo;
          for (int cc = 0; cc < 32; ++cc) {
            float g = 0.f;
#pragma unroll
            for (int v = 0; v < VDI; ++v) g += gtmpL[e][ol][v] * featL[e][cc][v];
            gRb[e][ol][cc] = g;
          }
        }
        for (int cc = hf * 16; cc < hf * 16 + 16; ++cc) {
          float gv[VDI];
#pragma unroll
          for (int v = 0; v < VDI; ++v) gv[v] = 0.f;
          for (int ol = 0; ol < 8; ++ol) {
#pragma unroll
            for (int v = 0; v < VDI; ++v) gv[v] += gtmpL[e][ol][v] * Rb[e][ol][cc];
          }
#pragma unroll
          for (int v = 0; v < VDI; ++v) gfeatL[e][cc][v] += gv[v];
        }
      }
      __syncthreads();
      // g_h2 += g_R @ W3^T  (accumulating WMMA chain over K=32 chunks)
      for (int q = 0; q < 8; ++q) {
        v16bf a2;
        const int e = lane & 15;
#pragma unroll
        for (int i = 0; i < 16; ++i) {
          int vg = i >> 1, hfb = i & 1;
          int K = ((vg & 3) * 2 + hfb) + ((lane >> 4) ? 8 : 0) + ((vg >> 2) ? 16 : 0);
          a2[i] = (__bf16)gRb[e][q][K];
        }
        const int gcol0 = f * 1024 + ob * 256 + q * 32;
        v16bf bl, bh;
        {
          const __bf16* wc = w3t + (size_t)gcol0 * HD + (lane & 15);
#pragma unroll
          for (int i = 0; i < 16; ++i) { int K = (i & 7) + ((lane >> 4) << 3) + ((i >> 3) << 4); bl[i] = wc[K * HD]; bh[i] = wc[K * HD + 16]; }
        }
        cg0 = __builtin_amdgcn_wmma_f32_16x16x32_bf16(false, a2, false, bl, (short)0, cg0, false, false);
        cg1 = __builtin_amdgcn_wmma_f32_16x16x32_bf16(false, a2, false, bh, (short)0, cg1, false, false);
      }
      __syncthreads();
    }  // ob
    if (PT != 0) {
      const int e = lane & 15;
#pragma unroll
      for (int k = 0; k < UDO * VDI; ++k) atomicAdd(&gbasL[e][k * NF + f], pb[k]);
    }
  }  // f
  __syncthreads();
  {
#pragma unroll
    for (int i = 0; i < 8; ++i) {
      int er = i + ((lane >> 4) << 3);
      gh2[(size_t)(e0 + er) * HD + (lane & 15)] = cg0[i];
      gh2[(size_t)(e0 + er) * HD + 16 + (lane & 15)] = cg1[i];
    }
  }
  for (int i = lane; i < 16 * 32 * VDI; i += 32) {
    int e = i / (32 * VDI), rm = i - e * 32 * VDI;
    atomicAdd(&ghin[(size_t)src[e0 + e] * (32 * VDI) + rm], gfeatL[e][rm / VDI][rm % VDI]);
  }
  if (PT != 0 && lane < 16 && gbas) {
    const int S = UDO * VDI * NF;
    for (int k = 0; k < S; ++k) gbas[(size_t)(e0 + lane) * S + k] += gbasL[lane][k];
  }
}

// ---------------------------- geometry backward -----------------------------
__global__ void k_geom_bwd(const int* src, const int* dst, const float* r,
                           const float* dirs, const float* g_r, const float* gb01,
                           const float* gb10, const float* gb11, float* g_pos) {
  int e = blockIdx.x * blockDim.x + threadIdx.x;
  if (e >= NE) return;
  const int AX[3] = {1, 2, 0};
  float dx = dirs[e * 3 + 0], dy = dirs[e * 3 + 1], dz = dirs[e * 3 + 2];
  float gY1[3] = {0.f, 0.f, 0.f};
  float gY2[5] = {0.f, 0.f, 0.f, 0.f, 0.f};
  for (int u = 0; u < 3; ++u) gY1[u] += gb01[e * 3 + u];
  for (int v = 0; v < 3; ++v) gY1[v] += C10 * gb10[e * 3 + v];
  const float* gb = gb11 + (size_t)e * 27;
  float gS[3][3] = {{0.f, 0.f, 0.f}, {0.f, 0.f, 0.f}, {0.f, 0.f, 0.f}};
  for (int u = 0; u < 3; ++u)
    for (int v = 0; v < 3; ++v) {
      float g1 = gb[(u * 3 + v) * 3 + 1];
      for (int j = 0; j < 3; ++j) gY1[j] += E111 * (float)epsv(AX[v], AX[j], AX[u]) * g1;
      gS[AX[u]][AX[v]] += gb[(u * 3 + v) * 3 + 2];
    }
  gY2[4] += 0.5f * KSA * (gS[0][0] - gS[1][1]);
  gY2[2] += KSB * (2.f * gS[2][2] - gS[0][0] - gS[1][1]);
  gY2[0] += KSC * (gS[0][1] + gS[1][0]);
  gY2[1] += KSC * (gS[1][2] + gS[2][1]);
  gY2[3] += KSC * (gS[0][2] + gS[2][0]);
  float gx = CY1 * gY1[2], gy = CY1 * gY1[0], gz = CY1 * gY1[1];
  gx += C2A * dy * gY2[0] + C2A * dz * gY2[3] + 2.f * C2C * dx * gY2[4];
  gy += C2A * dx * gY2[0] + C2A * dz * gY2[1] - 2.f * C2C * dy * gY2[4];
  gz += C2A * dy * gY2[1] + 6.f * C2B * dz * gY2[2] + C2A * dx * gY2[3];
  float gr = g_r[e], rr = r[e];
  float dotg = dx * gx + dy * gy + dz * gz;
  float gvx = gr * dx + (gx - dx * dotg) / rr;
  float gvy = gr * dy + (gy - dy * dotg) / rr;
  float gvz = gr * dz + (gz - dz * dotg) / rr;
  int s = src[e], d = dst[e];
  atomicAdd(&g_pos[d * 3 + 0], gvx); atomicAdd(&g_pos[d * 3 + 1], gvy);
  atomicAdd(&g_pos[d * 3 + 2], gvz);
  atomicAdd(&g_pos[s * 3 + 0], -gvx); atomicAdd(&g_pos[s * 3 + 1], -gvy);
  atomicAdd(&g_pos[s * 3 + 2], -gvz);
}

__global__ void k_finalize(const float* total, const float* g_pos, float* out) {
  int i = blockIdx.x * blockDim.x + threadIdx.x;
  if (i >= NG + NN * 3) return;
  out[i] = (i < NG) ? total[i] : -g_pos[i - NG];
}

// ------------------------------- host side ----------------------------------
struct PairH {
  const float *W1, *W2, *W3, *b1, *b2, *b3;
  int pt, nf, vdi, udo;
  float *h1, *h2, *tmp, *gh2;
  __bf16 *h2bf, *w3bf, *w3t;
};

static void msg_fwd_launch(const PairH& p, const int* src, const int* dst,
                           const float* dirs, const float* hin, float* acc,
                           hipStream_t s) {
  switch (p.pt) {
    case 0: k_msg_fwd<1,1,1,0><<<NE/16,32,0,s>>>(src,dst,dirs,hin,p.h2bf,p.w3bf,p.b3,p.tmp,acc); break;
    case 1: k_msg_fwd<1,1,3,1><<<NE/16,32,0,s>>>(src,dst,dirs,hin,p.h2bf,p.w3bf,p.b3,p.tmp,acc); break;
    case 2: k_msg_fwd<1,3,1,2><<<NE/16,32,0,s>>>(src,dst,dirs,hin,p.h2bf,p.w3bf,p.b3,p.tmp,acc); break;
    default: k_msg_fwd<3,3,3,3><<<NE/16,32,0,s>>>(src,dst,dirs,hin,p.h2bf,p.w3bf,p.b3,p.tmp,acc); break;
  }
}

static void msg_bwd_launch(const PairH& p, const int* src, const int* dst,
                           const float* dirs, const float* hin, const float* gacc,
                           float* ghin, float* gbas, hipStream_t s) {
  switch (p.pt) {
    case 0: k_msg_bwd<1,1,1,0><<<NE/16,32,0,s>>>(src,dst,dirs,hin,p.h2bf,p.w3bf,p.w3t,p.b3,p.tmp,gacc,ghin,p.gh2,gbas); break;
    case 1: k_msg_bwd<1,1,3,1><<<NE/16,32,0,s>>>(src,dst,dirs,hin,p.h2bf,p.w3bf,p.w3t,p.b3,p.tmp,gacc,ghin,p.gh2,gbas); break;
    case 2: k_msg_bwd<1,3,1,2><<<NE/16,32,0,s>>>(src,dst,dirs,hin,p.h2bf,p.w3bf,p.w3t,p.b3,p.tmp,gacc,ghin,p.gh2,gbas); break;
    default: k_msg_bwd<3,3,3,3><<<NE/16,32,0,s>>>(src,dst,dirs,hin,p.h2bf,p.w3bf,p.w3t,p.b3,p.tmp,gacc,ghin,p.gh2,gbas); break;
  }
}

extern "C" void kernel_launch(void* const* d_in, const int* in_sizes, int n_in,
                              void* d_out, int out_size, void* d_ws, size_t ws_size,
                              hipStream_t stream) {
  (void)in_sizes; (void)out_size;
  if (n_in < 88) return;
  auto F = [&](int i) { return (const float*)d_in[i]; };
  const int* nodef = (const int*)d_in[0];
  const int* src   = (const int*)d_in[1];
  const int* dst   = (const int*)d_in[2];
  const float* pos = (const float*)d_in[3];
  const int* gid   = (const int*)d_in[4];
  const float* emb = F(5);
  float* out = (float*)d_out;

  char* base = (char*)d_ws; size_t off = 0;
  auto alloc = [&](size_t bytes) -> char* {
    char* p = base + off; off += (bytes + 255) & ~(size_t)255; return p;
  };

  float* r    = (float*)alloc((size_t)NE * 4);
  float* dirs = (float*)alloc((size_t)NE * 12);
  float* cnt  = (float*)alloc((size_t)NN * 4);
  float* invd = (float*)alloc((size_t)NN * 4);
  float* h0   = (float*)alloc((size_t)NN * HD * 4);
  float* hA0 = (float*)alloc((size_t)NN * HD * 4);
  float* hA1 = (float*)alloc((size_t)NN * HD * 12);
  float* hB0 = (float*)alloc((size_t)NN * HD * 4);
  float* hB1 = (float*)alloc((size_t)NN * HD * 12);
  float* hC0 = (float*)alloc((size_t)NN * HD * 4);
  float* hC1 = (float*)alloc((size_t)NN * HD * 12);
  float* hD0 = (float*)alloc((size_t)NN * HD * 4);
  float* accA0 = (float*)alloc((size_t)NN * HD * 4);
  float* accA1 = (float*)alloc((size_t)NN * HD * 12);
  float* accB0 = (float*)alloc((size_t)NN * HD * 4);
  float* accB1 = (float*)alloc((size_t)NN * HD * 12);
  float* accC0 = (float*)alloc((size_t)NN * HD * 4);
  float* accC1 = (float*)alloc((size_t)NN * HD * 12);
  float* accD0 = (float*)alloc((size_t)NN * HD * 4);
  float* gaccA0 = (float*)alloc((size_t)NN * HD * 4);
  float* gaccA1 = (float*)alloc((size_t)NN * HD * 12);
  float* gaccB0 = (float*)alloc((size_t)NN * HD * 4);
  float* gaccB1 = (float*)alloc((size_t)NN * HD * 12);
  float* gaccC0 = (float*)alloc((size_t)NN * HD * 4);
  float* gaccC1 = (float*)alloc((size_t)NN * HD * 12);
  float* gaccD0 = (float*)alloc((size_t)NN * HD * 4);
  float* gh0d = (float*)alloc((size_t)NN * HD * 12);
  float* ghA0 = (float*)alloc((size_t)NN * HD * 4);
  float* ghA1 = (float*)alloc((size_t)NN * HD * 12);
  float* ghB0 = (float*)alloc((size_t)NN * HD * 4);
  float* ghB1 = (float*)alloc((size_t)NN * HD * 12);
  float* ghC0 = (float*)alloc((size_t)NN * HD * 4);
  float* ghC1 = (float*)alloc((size_t)NN * HD * 12);
  float* ghD0 = (float*)alloc((size_t)NN * HD * 4);
  float* total = (float*)alloc(64);
  float* g_pos = (float*)alloc((size_t)NN * 12);
  float* g_r   = (float*)alloc((size_t)NE * 4);
  float* gb01  = (float*)alloc((size_t)NE * 12);
  float* gb10  = (float*)alloc((size_t)NE * 12);
  float* gb11  = (float*)alloc((size_t)NE * 108);

  // pairs: 0-1:L0{(0,0),(0,1)}  2-5:L1{(0,0),(0,1),(1,0),(1,1)}  6-9:L2  10-11:final{(0,0),(1,0)}
  PairH P[12];
  const int rbase[12] = {19, 25, 32, 38, 44, 50, 58, 64, 70, 76, 6, 12};
  const int ptab[12]  = {0, 1, 0, 1, 2, 3, 0, 1, 2, 3, 0, 2};
  for (int p = 0; p < 12; ++p) {
    int rb = rbase[p];
    P[p].W1 = F(rb); P[p].W2 = F(rb + 1); P[p].W3 = F(rb + 2);
    P[p].b1 = F(rb + 3); P[p].b2 = F(rb + 4); P[p].b3 = F(rb + 5);
    P[p].pt = ptab[p];
    P[p].nf  = (ptab[p] == 3) ? 3 : 1;
    P[p].vdi = (ptab[p] >= 2) ? 3 : 1;
    P[p].udo = (ptab[p] == 1 || ptab[p] == 3) ? 3 : 1;
    P[p].h1  = (float*)alloc((size_t)NE * HD * 4);
    P[p].h2  = (float*)alloc((size_t)NE * HD * 4);
    P[p].gh2 = (float*)alloc((size_t)NE * HD * 4);
    P[p].h2bf = (__bf16*)alloc((size_t)NE * HD * 2);
    int cols = P[p].nf * 1024;
    P[p].w3bf = (__bf16*)alloc((size_t)32 * cols * 2);
    P[p].w3t  = (__bf16*)alloc((size_t)cols * 32 * 2);
    P[p].tmp  = (float*)alloc((size_t)NE * P[p].nf * 32 * P[p].vdi * 4);
  }
  if (off > ws_size) return;

  const float* s0L0 = F(31);
  const float* s0L1 = F(56); const float* s1L1 = F(57);
  const float* s0L2 = F(82); const float* s1L2 = F(83);
  const float* s0F  = F(18);
  const float* roW1 = F(84); const float* roW2 = F(85);
  const float* rob1 = F(86); const float* rob2 = F(87);

  // zero accumulators
  hipMemsetAsync(cnt, 0, (size_t)NN * 4, stream);
  hipMemsetAsync(total, 0, 64, stream);
  hipMemsetAsync(accA0, 0, (size_t)NN * HD * 4, stream);
  hipMemsetAsync(accA1, 0, (size_t)NN * HD * 12, stream);
  hipMemsetAsync(accB0, 0, (size_t)NN * HD * 4, stream);
  hipMemsetAsync(accB1, 0, (size_t)NN * HD * 12, stream);
  hipMemsetAsync(accC0, 0, (size_t)NN * HD * 4, stream);
  hipMemsetAsync(accC1, 0, (size_t)NN * HD * 12, stream);
  hipMemsetAsync(accD0, 0, (size_t)NN * HD * 4, stream);
  hipMemsetAsync(gh0d, 0, (size_t)NN * HD * 12, stream);
  hipMemsetAsync(ghA0, 0, (size_t)NN * HD * 4, stream);
  hipMemsetAsync(ghA1, 0, (size_t)NN * HD * 12, stream);
  hipMemsetAsync(ghB0, 0, (size_t)NN * HD * 4, stream);
  hipMemsetAsync(ghB1, 0, (size_t)NN * HD * 12, stream);
  hipMemsetAsync(ghC0, 0, (size_t)NN * HD * 4, stream);
  hipMemsetAsync(ghC1, 0, (size_t)NN * HD * 12, stream);
  hipMemsetAsync(g_pos, 0, (size_t)NN * 12, stream);
  hipMemsetAsync(g_r, 0, (size_t)NE * 4, stream);
  hipMemsetAsync(gb01, 0, (size_t)NE * 12, stream);
  hipMemsetAsync(gb10, 0, (size_t)NE * 12, stream);
  hipMemsetAsync(gb11, 0, (size_t)NE * 108, stream);

  // -------- forward --------
  k_geom<<<(NE + 255) / 256, 256, 0, stream>>>(pos, src, dst, r, dirs, cnt);
  k_invdeg<<<(NN + 255) / 256, 256, 0, stream>>>(cnt, invd);
  k_embed<<<NN * HD / 256, 256, 0, stream>>>(nodef, emb, h0);
  for (int p = 0; p < 12; ++p) {
    int cols = P[p].nf * 1024;
    k_w3prep<<<(32 * cols + 255) / 256, 256, 0, stream>>>(P[p].W3, P[p].w3bf, P[p].w3t, cols);
    k_radial_fwd<<<NE / 8, 256, 0, stream>>>(r, P[p].W1, P[p].b1, P[p].W2, P[p].b2,
                                             P[p].h1, P[p].h2, P[p].h2bf);
  }
  // L0
  msg_fwd_launch(P[0], src, dst, dirs, h0, accA0, stream);
  msg_fwd_launch(P[1], src, dst, dirs, h0, accA1, stream);
  k_node_upd<<<(NN * HD + 255) / 256, 256, 0, stream>>>(accA0, invd, s0L0, h0, hA0, 1);
  k_node_upd<<<(NN * HD * 3 + 255) / 256, 256, 0, stream>>>(accA1, invd, nullptr, nullptr, hA1, 3);
  // L1
  msg_fwd_launch(P[2], src, dst, dirs, hA0, accB0, stream);
  msg_fwd_launch(P[3], src, dst, dirs, hA0, accB1, stream);
  msg_fwd_launch(P[4], src, dst, dirs, hA1, accB0, stream);
  msg_fwd_launch(P[5], src, dst, dirs, hA1, accB1, stream);
  k_node_upd<<<(NN * HD + 255) / 256, 256, 0, stream>>>(accB0, invd, s0L1, hA0, hB0, 1);
  k_node_upd<<<(NN * HD * 3 + 255) / 256, 256, 0, stream>>>(accB1, invd, s1L1, hA1, hB1, 3);
  // L2
  msg_fwd_launch(P[6], src, dst, dirs, hB0, accC0, stream);
  msg_fwd_launch(P[7], src, dst, dirs, hB0, accC1, stream);
  msg_fwd_launch(P[8], src, dst, dirs, hB1, accC0, stream);
  msg_fwd_launch(P[9], src, dst, dirs, hB1, accC1, stream);
  k_node_upd<<<(NN * HD + 255) / 256, 256, 0, stream>>>(accC0, invd, s0L2, hB0, hC0, 1);
  k_node_upd<<<(NN * HD * 3 + 255) / 256, 256, 0, stream>>>(accC1, invd, s1L2, hB1, hC1, 3);
  // final
  msg_fwd_launch(P[10], src, dst, dirs, hC0, accD0, stream);
  msg_fwd_launch(P[11], src, dst, dirs, hC1, accD0, stream);
  k_node_upd<<<(NN * HD + 255) / 256, 256, 0, stream>>>(accD0, invd, s0F, hC0, hD0, 1);
  k_readout_fwd<<<NN, 32, 0, stream>>>(hD0, roW1, rob1, roW2, rob2, gid, total);

  // -------- backward --------
  k_readout_bwd<<<NN, 32, 0, stream>>>(hD0, roW1, rob1, roW2, ghD0);
  // final layer
  k_node_upd_bwd<<<(NN * HD + 255) / 256, 256, 0, stream>>>(ghD0, invd, s0F, gaccD0, ghC0, 1);
  msg_bwd_launch(P[10], src, dst, dirs, hC0, gaccD0, ghC0, nullptr, stream);
  msg_bwd_launch(P[11], src, dst, dirs, hC1, gaccD0, ghC1, gb10, stream);
  k_radial_bwd<<<NE / 8, 256, 0, stream>>>(P[10].gh2, P[10].h1, P[10].h2, P[10].W1, P[10].W2, g_r);
  k_radial_bwd<<<NE / 8, 256, 0, stream>>>(P[11].gh2, P[11].h1, P[11].h2, P[11].W1, P[11].W2, g_r);
  // L2
  k_node_upd_bwd<<<(NN * HD + 255) / 256, 256, 0, stream>>>(ghC0, invd, s0L2, gaccC0, ghB0, 1);
  k_node_upd_bwd<<<(NN * HD * 3 + 255) / 256, 256, 0, stream>>>(ghC1, invd, s1L2, gaccC1, ghB1, 3);
  msg_bwd_launch(P[6], src, dst, dirs, hB0, gaccC0, ghB0, nullptr, stream);
  msg_bwd_launch(P[7], src, dst, dirs, hB0, gaccC1, ghB0, gb01, stream);
  msg_bwd_launch(P[8], src, dst, dirs, hB1, gaccC0, ghB1, gb10, stream);
  msg_bwd_launch(P[9], src, dst, dirs, hB1, gaccC1, ghB1, gb11, stream);
  for (int p = 6; p <= 9; ++p)
    k_radial_bwd<<<NE / 8, 256, 0, stream>>>(P[p].gh2, P[p].h1, P[p].h2, P[p].W1, P[p].W2, g_r);
  // L1
  k_node_upd_bwd<<<(NN * HD + 255) / 256, 256, 0, stream>>>(ghB0, invd, s0L1, gaccB0, ghA0, 1);
  k_node_upd_bwd<<<(NN * HD * 3 + 255) / 256, 256, 0, stream>>>(ghB1, invd, s1L1, gaccB1, ghA1, 3);
  msg_bwd_launch(P[2], src, dst, dirs, hA0, gaccB0, ghA0, nullptr, stream);
  msg_bwd_launch(P[3], src, dst, dirs, hA0, gaccB1, ghA0, gb01, stream);
  msg_bwd_launch(P[4], src, dst, dirs, hA1, gaccB0, ghA1, gb10, stream);
  msg_bwd_launch(P[5], src, dst, dirs, hA1, gaccB1, ghA1, gb11, stream);
  for (int p = 2; p <= 5; ++p)
    k_radial_bwd<<<NE / 8, 256, 0, stream>>>(P[p].gh2, P[p].h1, P[p].h2, P[p].W1, P[p].W2, g_r);
  // L0 (feature grads stop at embedding; still need g_h2 / g_basis)
  k_node_upd_bwd<<<(NN * HD + 255) / 256, 256, 0, stream>>>(ghA0, invd, s0L0, gaccA0, gh0d, 1);
  k_node_upd_bwd<<<(NN * HD * 3 + 255) / 256, 256, 0, stream>>>(ghA1, invd, nullptr, gaccA1, nullptr, 3);
  msg_bwd_launch(P[0], src, dst, dirs, h0, gaccA0, gh0d, nullptr, stream);
  msg_bwd_launch(P[1], src, dst, dirs, h0, gaccA1, gh0d, gb01, stream);
  k_radial_bwd<<<NE / 8, 256, 0, stream>>>(P[0].gh2, P[0].h1, P[0].h2, P[0].W1, P[0].W2, g_r);
  k_radial_bwd<<<NE / 8, 256, 0, stream>>>(P[1].gh2, P[1].h1, P[1].h2, P[1].W1, P[1].W2, g_r);
  // geometry -> forces
  k_geom_bwd<<<(NE + 255) / 256, 256, 0, stream>>>(src, dst, r, dirs, g_r, gb01, gb10, gb11, g_pos);
  k_finalize<<<(NG + NN * 3 + 255) / 256, 256, 0, stream>>>(total, g_pos, out);
}
